// PointerSelectorV4_9_46780783788507
// MI455X (gfx1250) — matrix operand order repaired and checked by reading, not compiled, not run on hardware
//
#include <hip/hip_runtime.h>

typedef __attribute__((ext_vector_type(16))) _Float16 v16h;
typedef __attribute__((ext_vector_type(8)))  _Float16 v8h;
typedef __attribute__((ext_vector_type(8)))  float    v8f;
typedef _Float16 halft;

union FragAB { v16h v; v8h h[2]; };

// ---------------- wave reductions (wave32) ----------------
__device__ inline float wsum(float v) {
#pragma unroll
  for (int o = 16; o; o >>= 1) v += __shfl_xor(v, o, 32);
  return v;
}
__device__ inline float wmaxr(float v) {
#pragma unroll
  for (int o = 16; o; o >>= 1) v = fmaxf(v, __shfl_xor(v, o, 32));
  return v;
}

__device__ inline v8h cvt8(float4 a, float4 b) {
  v8h r;
  r[0] = (_Float16)a.x; r[1] = (_Float16)a.y; r[2] = (_Float16)a.z; r[3] = (_Float16)a.w;
  r[4] = (_Float16)b.x; r[5] = (_Float16)b.y; r[6] = (_Float16)b.z; r[7] = (_Float16)b.w;
  return r;
}

// load 16 elements starting at (row, koff) and return them as 2 x v8h (f16)
__device__ inline void loadcvt(const float* __restrict__ base, int row, int valid,
                               int ld, int koff, v8h out[2]) {
  if (valid) {
    const float4* p = reinterpret_cast<const float4*>(base + (size_t)row * ld + koff);
    float4 r0 = p[0], r1 = p[1], r2 = p[2], r3 = p[3];
    out[0] = cvt8(r0, r1);
    out[1] = cvt8(r2, r3);
  } else {
    v8h z;
#pragma unroll
    for (int i = 0; i < 8; ++i) z[i] = (_Float16)0.f;
    out[0] = out[1] = z;
  }
}
__device__ inline void loadcvt(const halft* __restrict__ base, int row, int valid,
                               int ld, int koff, v8h out[2]) {
  if (valid) {
    const v8h* p = reinterpret_cast<const v8h*>(base + (size_t)row * ld + koff);
    out[0] = p[0];
    out[1] = p[1];
  } else {
    v8h z;
#pragma unroll
    for (int i = 0; i < 8; ++i) z[i] = (_Float16)0.f;
    out[0] = out[1] = z;
  }
}

__device__ inline void store_out(float* __restrict__ Y, size_t idx, float v) { Y[idx] = v; }
__device__ inline void store_out(halft* __restrict__ Y, size_t idx, float v) { Y[idx] = (_Float16)v; }

// ======================================================================
// Generic WMMA GEMM:  Y[N,Hout] = X[N,Kd] @ W[Hout,Kd]^T (+ bias)
// TX in {float, f16}: f32 inputs converted in-register; f16 inputs staged
// straight to LDS. TY in {float, f16}. f32 accumulation via
// v_wmma_f32_16x16x32_f16.
// Block: 256 thr (8 wave32); block tile 128x128; wave tile 32x64; K-step 32.
// Double-buffered LDS, 1 barrier / k-step.
// Requires: Kd % 32 == 0, Hout % 128 == 0 (true for all uses: 256/768).
// ======================================================================
#define BM 128
#define BN 128
#define BKK 32
#define LDSS 40  // halfs per LDS row (32 + 8 pad)

template <typename TX, typename TY>
__global__ __launch_bounds__(256) void wmma_gemm_kernel(
    const TX* __restrict__ X, const float* __restrict__ W,
    const float* __restrict__ bias, TY* __restrict__ Y,
    int N, int Kd, int Hout) {
  __shared__ __align__(16) _Float16 Asub[2][BM][LDSS];
  __shared__ __align__(16) _Float16 Bsub[2][BN][LDSS];

  const int tid  = threadIdx.x;
  const int lane = tid & 31;
  const int wid  = tid >> 5;
  const int wr   = wid & 3;      // 0..3 : wave row
  const int wc   = wid >> 2;     // 0..1 : wave col
  const int half = lane >> 4;
  const int m16  = lane & 15;

  const int rowBase = blockIdx.x * BM;
  const int colBase = blockIdx.y * BN;

  const int ldRow = tid >> 1;         // 0..127
  const int ldSeg = (tid & 1) * 16;   // 0 or 16 (elements)

  v8f acc[2][4];
  const v8f vz = {0.f, 0.f, 0.f, 0.f, 0.f, 0.f, 0.f, 0.f};
#pragma unroll
  for (int r = 0; r < 2; ++r)
#pragma unroll
    for (int c = 0; c < 4; ++c) acc[r][c] = vz;

  const int nk = Kd / BKK;
  const int xrow = rowBase + ldRow;
  const int wrow = colBase + ldRow;

  // prologue: stage tile 0
  {
    v8h ah[2], bh[2];
    loadcvt(X, xrow, xrow < N, Kd, ldSeg, ah);
    loadcvt(W, wrow, 1, Kd, ldSeg, bh);
    *(v8h*)&Asub[0][ldRow][ldSeg]     = ah[0];
    *(v8h*)&Asub[0][ldRow][ldSeg + 8] = ah[1];
    *(v8h*)&Bsub[0][ldRow][ldSeg]     = bh[0];
    *(v8h*)&Bsub[0][ldRow][ldSeg + 8] = bh[1];
  }

  int cur = 0;
  for (int kc = 0; kc < nk; ++kc) {
    __syncthreads();
    v8h ah[2], bh[2];
    const bool hasNext = (kc + 1 < nk);
    if (hasNext) {
      const int kk = (kc + 1) * BKK + ldSeg;
      loadcvt(X, xrow, xrow < N, Kd, kk, ah);
      loadcvt(W, wrow, 1, Kd, kk, bh);
      if (kc + 2 < nk && xrow < N)
        __builtin_prefetch(X + (size_t)xrow * Kd + (kc + 2) * BKK, 0, 0);
    }

    FragAB fa[2], fb[4];
#pragma unroll
    for (int r = 0; r < 2; ++r) {
      const int arow = wr * 32 + r * 16 + m16;
      fa[r].h[0] = *(const v8h*)&Asub[cur][arow][half * 8];
      fa[r].h[1] = *(const v8h*)&Asub[cur][arow][16 + half * 8];
    }
#pragma unroll
    for (int c = 0; c < 4; ++c) {
      const int bcol = wc * 64 + c * 16 + m16;
      fb[c].h[0] = *(const v8h*)&Bsub[cur][bcol][half * 8];
      fb[c].h[1] = *(const v8h*)&Bsub[cur][bcol][16 + half * 8];
    }
#pragma unroll
    for (int r = 0; r < 2; ++r)
#pragma unroll
      for (int c = 0; c < 4; ++c)
        acc[r][c] = __builtin_amdgcn_wmma_f32_16x16x32_f16(
            false, fa[r].v, false, fb[c].v, (short)0, acc[r][c], false, false);

    if (hasNext) {
      const int nxt = cur ^ 1;
      *(v8h*)&Asub[nxt][ldRow][ldSeg]     = ah[0];
      *(v8h*)&Asub[nxt][ldRow][ldSeg + 8] = ah[1];
      *(v8h*)&Bsub[nxt][ldRow][ldSeg]     = bh[0];
      *(v8h*)&Bsub[nxt][ldRow][ldSeg + 8] = bh[1];
    }
    cur ^= 1;
  }

  // epilogue: C/D layout: lane -> n = m16, rows m = half*8 + vgpr
#pragma unroll
  for (int r = 0; r < 2; ++r) {
#pragma unroll
    for (int c = 0; c < 4; ++c) {
      const int col = colBase + wc * 64 + c * 16 + m16;
      const float bv = bias ? bias[col] : 0.0f;
#pragma unroll
      for (int v = 0; v < 8; ++v) {
        const int row = rowBase + wr * 32 + r * 16 + half * 8 + v;
        if (row < N) store_out(Y, (size_t)row * Hout + col, acc[r][c][v] + bv);
      }
    }
  }
}

// ======================================================================
// Attention: per batch row, softmax over kc keys (kc = 32 or 8), H = 256.
// One wave per batch row; K/V are f16 (WMMA GEMM outputs), q/ctx f32.
// ======================================================================
__global__ __launch_bounds__(256) void attn_kernel(
    const float* __restrict__ qp, const halft* __restrict__ kp,
    const halft* __restrict__ vp, float* __restrict__ ctx,
    int b0, int kc, float scale) {
  __shared__ __align__(16) float qs[8][256];
  const int lane = threadIdx.x & 31;
  const int wid  = threadIdx.x >> 5;
  const int bl   = blockIdx.x * 8 + wid;   // chunk-local
  const int b    = b0 + bl;                // global batch

  {
    const float4* qrow = (const float4*)(qp + (size_t)b * 256);
    *(float4*)&qs[wid][lane * 8]     = qrow[lane * 2];
    *(float4*)&qs[wid][lane * 8 + 4] = qrow[lane * 2 + 1];
  }
  __syncthreads();

  float s = -1e30f;
  if (lane < kc) {
    const v8h* kr = (const v8h*)(kp + ((size_t)bl * kc + lane) * 256);
    const float* qv = qs[wid];
    float d = 0.f;
#pragma unroll 4
    for (int j = 0; j < 32; ++j) {
      v8h k8 = kr[j];
#pragma unroll
      for (int t = 0; t < 8; ++t) d += qv[j * 8 + t] * (float)k8[t];
    }
    s = d * scale;
  }
  const float m = wmaxr(s);
  const float e = (lane < kc) ? __expf(s - m) : 0.f;
  const float den = wsum(e);
  const float a = e / den;

  float acc0[8];
#pragma unroll
  for (int j = 0; j < 8; ++j) acc0[j] = 0.f;
  for (int k2 = 0; k2 < kc; ++k2) {
    const float ak = __shfl(a, k2, 32);
    const v8h* vr = (const v8h*)(vp + ((size_t)bl * kc + k2) * 256 + lane * 8);
    v8h v0 = vr[0];
#pragma unroll
    for (int j = 0; j < 8; ++j) acc0[j] += ak * (float)v0[j];
  }
  float* o = ctx + (size_t)b * 256 + lane * 8;
#pragma unroll
  for (int j = 0; j < 8; ++j) o[j] = acc0[j];
}

// residual + LayerNorm over rows of 256 (one wave per row)
__global__ __launch_bounds__(256) void resln_kernel(
    const float* __restrict__ a, const float* __restrict__ res,
    const float* __restrict__ g, const float* __restrict__ bta,
    float* __restrict__ out) {
  const int lane = threadIdx.x & 31;
  const int wid  = threadIdx.x >> 5;
  const size_t row = (size_t)blockIdx.x * 8 + wid;
  float x[8];
  float s = 0.f;
#pragma unroll
  for (int j = 0; j < 8; ++j) {
    x[j] = a[row * 256 + lane * 8 + j] + res[row * 256 + lane * 8 + j];
    s += x[j];
  }
  const float mean = wsum(s) * (1.f / 256.f);
  float v = 0.f;
#pragma unroll
  for (int j = 0; j < 8; ++j) { const float d = x[j] - mean; v += d * d; }
  const float inv = rsqrtf(wsum(v) * (1.f / 256.f) + 1e-5f);
#pragma unroll
  for (int j = 0; j < 8; ++j)
    out[row * 256 + lane * 8 + j] =
        (x[j] - mean) * inv * g[lane * 8 + j] + bta[lane * 8 + j];
}

// in-place row L2 normalize (rows of 256)
__global__ __launch_bounds__(256) void l2n_kernel(float* __restrict__ x) {
  const int lane = threadIdx.x & 31;
  const int wid  = threadIdx.x >> 5;
  const size_t row = (size_t)blockIdx.x * 8 + wid;
  float v[8];
  float ss = 0.f;
#pragma unroll
  for (int j = 0; j < 8; ++j) { v[j] = x[row * 256 + lane * 8 + j]; ss += v[j] * v[j]; }
  const float inv = 1.f / fmaxf(sqrtf(wsum(ss)), 1e-12f);
#pragma unroll
  for (int j = 0; j < 8; ++j) x[row * 256 + lane * 8 + j] = v[j] * inv;
}

// cheap scores + top-8 + gather cand_sub (f16) ; also hs = l2n(h + step_emb)
__global__ __launch_bounds__(256) void cheap_topk_kernel(
    const float* __restrict__ h, const float* __restrict__ se,
    const float* __restrict__ cn, const float* __restrict__ mask,
    float* __restrict__ hs, int* __restrict__ tidx,
    halft* __restrict__ csub, int step) {
  __shared__ __align__(16) float hss[8][256];
  const int lane = threadIdx.x & 31;
  const int wid  = threadIdx.x >> 5;
  const size_t b = (size_t)blockIdx.x * 8 + wid;

  float x[8];
  float ss = 0.f;
#pragma unroll
  for (int j = 0; j < 8; ++j) {
    x[j] = h[b * 256 + lane * 8 + j] + se[(size_t)step * 256 + lane * 8 + j];
    ss += x[j] * x[j];
  }
  const float inv = 1.f / fmaxf(sqrtf(wsum(ss)), 1e-12f);
#pragma unroll
  for (int j = 0; j < 8; ++j) {
    x[j] *= inv;
    hss[wid][lane * 8 + j] = x[j];
    hs[b * 256 + lane * 8 + j] = x[j];
  }
  __syncthreads();

  const float4* cr4 = (const float4*)(cn + (b * 32 + lane) * 256);
  const float4* q4  = (const float4*)hss[wid];
  float d = 0.f;
#pragma unroll 8
  for (int j = 0; j < 64; ++j) {
    float4 a = q4[j], c4 = cr4[j];
    d += a.x * c4.x + a.y * c4.y + a.z * c4.z + a.w * c4.w;
  }
  float val = (mask[b * 32 + lane] != 0.f) ? -100.f : d * 10.f;  // /TEMP

  int idxs[8];
#pragma unroll
  for (int m = 0; m < 8; ++m) {
    float bv = val; int bi = lane;
#pragma unroll
    for (int o = 16; o; o >>= 1) {
      const float ov = __shfl_xor(bv, o, 32);
      const int   oi = __shfl_xor(bi, o, 32);
      if (ov > bv || (ov == bv && oi < bi)) { bv = ov; bi = oi; }
    }
    idxs[m] = bi;
    if (lane == bi) val = -1e30f;
  }
  if (lane < 8) tidx[b * 8 + lane] = idxs[lane];
#pragma unroll
  for (int m = 0; m < 8; ++m) {
    const float* src = cn + (b * 32 + idxs[m]) * 256;
    halft* dst = csub + (b * 8 + m) * 256;
#pragma unroll
    for (int j = 0; j < 8; ++j) dst[lane * 8 + j] = (_Float16)src[lane * 8 + j];
  }
}

// refined scores -> scatter into logits, argmax, mask update, gather chosen
__global__ __launch_bounds__(256) void score_kernel(
    const float* __restrict__ rq, const halft* __restrict__ csub,
    const int* __restrict__ tidx, float* __restrict__ mask,
    const int* __restrict__ labels, const float* __restrict__ cn,
    float* __restrict__ logits, float* __restrict__ preds,
    float* __restrict__ chosen, int step) {
  const int lane = threadIdx.x & 31;
  const int wid  = threadIdx.x >> 5;
  const size_t b = (size_t)blockIdx.x * 8 + wid;

  float rv[8];
#pragma unroll
  for (int j = 0; j < 8; ++j) rv[j] = rq[b * 256 + lane * 8 + j];

  int idxm[8];
#pragma unroll
  for (int m = 0; m < 8; ++m) idxm[m] = tidx[b * 8 + m];

  float refined[8];
#pragma unroll
  for (int m = 0; m < 8; ++m) {
    const halft* cs = csub + (b * 8 + m) * 256;
    float d = 0.f, ss = 0.f;
#pragma unroll
    for (int j = 0; j < 8; ++j) {
      const float c = (float)cs[lane * 8 + j];
      d += rv[j] * c; ss += c * c;
    }
    d = wsum(d); ss = wsum(ss);
    refined[m] = d / fmaxf(sqrtf(ss), 1e-12f) * 10.f;  // /TEMP
  }

  float sc = -100.f;
#pragma unroll
  for (int m = 0; m < 8; ++m)
    if (idxm[m] == lane) sc = refined[m];
  if (mask[b * 32 + lane] != 0.f) sc = -100.f;
  logits[(b * 2 + step) * 32 + lane] = sc;

  float bv = sc; int bi = lane;
#pragma unroll
  for (int o = 16; o; o >>= 1) {
    const float ov = __shfl_xor(bv, o, 32);
    const int   oi = __shfl_xor(bi, o, 32);
    if (ov > bv || (ov == bv && oi < bi)) { bv = ov; bi = oi; }
  }
  const int lab = labels[b * 2 + step];
  if (lane == 0) {
    preds[b * 2 + step] = (float)bi;
    mask[b * 32 + lab] = 1.0f;
  }
  const float* src = cn + (b * 32 + lab) * 256;
#pragma unroll
  for (int j = 0; j < 8; ++j) chosen[b * 256 + lane * 8 + j] = src[lane * 8 + j];
}

// GRU elementwise + in-place L2 normalize of new h
__global__ __launch_bounds__(256) void gru_kernel(
    const float* __restrict__ gi, const float* __restrict__ gh,
    float* __restrict__ h) {
  const int lane = threadIdx.x & 31;
  const int wid  = threadIdx.x >> 5;
  const size_t b = (size_t)blockIdx.x * 8 + wid;
  float nh[8];
  float ss = 0.f;
#pragma unroll
  for (int j = 0; j < 8; ++j) {
    const int e = lane * 8 + j;
    const float ir = gi[b * 768 + e],       hr = gh[b * 768 + e];
    const float iz = gi[b * 768 + 256 + e], hz = gh[b * 768 + 256 + e];
    const float in = gi[b * 768 + 512 + e], hn = gh[b * 768 + 512 + e];
    const float hv = h[b * 256 + e];
    const float r = 1.f / (1.f + __expf(-(ir + hr)));
    const float z = 1.f / (1.f + __expf(-(iz + hz)));
    const float n = tanhf(in + r * hn);
    const float o = (1.f - z) * n + z * hv;
    nh[j] = o; ss += o * o;
  }
  const float inv = 1.f / fmaxf(sqrtf(wsum(ss)), 1e-12f);
#pragma unroll
  for (int j = 0; j < 8; ++j) h[b * 256 + lane * 8 + j] = nh[j] * inv;
}

// per-row label-smoothed NLL (rows = B*S, 32 logits per row)
__global__ __launch_bounds__(256) void rowloss_kernel(
    const float* __restrict__ logits, const int* __restrict__ labels,
    float* __restrict__ rl) {
  const int lane = threadIdx.x & 31;
  const int wid  = threadIdx.x >> 5;
  const size_t row = (size_t)blockIdx.x * 8 + wid;
  const float lf = fmaxf(logits[row * 32 + lane], -100.f);
  const float m = wmaxr(lf);
  const float e = __expf(lf - m);
  const float lse = logf(wsum(e)) + m;
  const float lp = lf - lse;
  const int y = labels[row];
  const float nll = wsum((lane == y) ? -lp : 0.f);
  const float mn = wsum(-lp) * (1.f / 32.f);
  if (lane == 0) rl[row] = 0.9f * nll + 0.1f * mn;
}

__global__ __launch_bounds__(256) void finloss_kernel(
    const float* __restrict__ rl, float* __restrict__ out) {
  __shared__ float red[256];
  const int t = threadIdx.x;
  float s = 0.f;
  for (int i = t; i < 16384; i += 256) s += rl[i];
  red[t] = s;
  __syncthreads();
  for (int o = 128; o; o >>= 1) {
    if (t < o) red[t] += red[t + o];
    __syncthreads();
  }
  if (t == 0) out[0] = red[0] * (1.f / 16384.f);
}

__global__ void zero_kernel(float* __restrict__ p, int n) {
  const int i = blockIdx.x * 256 + threadIdx.x;
  if (i < n) p[i] = 0.f;
}

// ======================================================================
// Host orchestration
// ======================================================================
template <typename TX, typename TY>
static inline void launch_gemm(const TX* X, const float* W, const float* bias,
                               TY* Y, int N, int Kd, int Hout, hipStream_t s) {
  dim3 g((N + BM - 1) / BM, Hout / BN), b(256);
  wmma_gemm_kernel<TX, TY><<<g, b, 0, s>>>(X, W, bias, Y, N, Kd, Hout);
}

extern "C" void kernel_launch(void* const* d_in, const int* in_sizes, int n_in,
                              void* d_out, int out_size, void* d_ws, size_t ws_size,
                              hipStream_t stream) {
  (void)in_sizes; (void)n_in; (void)out_size; (void)ws_size;
  constexpr int B = 8192, D = 768, H = 256, K = 32, S = 2, M = 8;
  constexpr int CH = 1024;  // attention chunk (batches)

  const float* query_emb = (const float*)d_in[0];
  const float* cand_emb  = (const float*)d_in[1];
  const int*   labels    = (const int*)d_in[2];
  const float* ipw       = (const float*)d_in[3];
  const float* ca_in_w   = (const float*)d_in[4];
  const float* ca_in_b   = (const float*)d_in[5];
  const float* ca_out_w  = (const float*)d_in[6];
  const float* ca_out_b  = (const float*)d_in[7];
  const float* ln1_g     = (const float*)d_in[8];
  const float* ln1_b     = (const float*)d_in[9];
  const float* qpw       = (const float*)d_in[10];
  const float* cpw       = (const float*)d_in[11];
  const float* rf_in_w   = (const float*)d_in[12];
  const float* rf_in_b   = (const float*)d_in[13];
  const float* rf_out_w  = (const float*)d_in[14];
  const float* rf_out_b  = (const float*)d_in[15];
  const float* rf_ln_g   = (const float*)d_in[16];
  const float* rf_ln_b   = (const float*)d_in[17];
  const float* gru_w_ih  = (const float*)d_in[18];
  const float* gru_b_ih  = (const float*)d_in[19];
  const float* gru_w_hh  = (const float*)d_in[20];
  const float* gru_b_hh  = (const float*)d_in[21];
  const float* step_emb  = (const float*)d_in[22];

  float* out    = (float*)d_out;
  float* logits = out;                       // B*S*K = 524288
  float* preds  = out + (size_t)B * S * K;   // B*S   = 16384
  float* lossp  = preds + (size_t)B * S;     // 1

  // ---- workspace layout (bump allocator, float units) ----
  float* ws = (float*)d_ws;
  size_t off = 0;
  auto alloc = [&](size_t nfloats) { float* p = ws + off; off += nfloats; return p; };
  halft* crh     = (halft*)alloc((size_t)B * K * H / 2);   // f16; reused after cn
  float* cn      = alloc((size_t)B * K * H);               // 67.1M f32
  float* qr      = alloc((size_t)B * H);
  float* qp      = alloc((size_t)B * H);
  float* ctx     = alloc((size_t)B * H);
  float* attn_out= alloc((size_t)B * H);
  float* qa      = alloc((size_t)B * H);
  float* hbuf    = alloc((size_t)B * H);
  float* hs      = alloc((size_t)B * H);
  float* qp2     = alloc((size_t)B * H);
  float* ctx2    = alloc((size_t)B * H);
  float* aout2   = alloc((size_t)B * H);
  float* rtmp    = alloc((size_t)B * H);
  float* chosen  = alloc((size_t)B * H);
  float* gi      = alloc((size_t)B * 3 * H);
  float* gh      = alloc((size_t)B * 3 * H);
  halft* kpch    = (halft*)alloc((size_t)CH * K * H / 2);  // f16 chunk K
  halft* vpch    = (halft*)alloc((size_t)CH * K * H / 2);  // f16 chunk V
  float* maskb   = alloc((size_t)B * K);
  float* rowloss = alloc((size_t)B * S);
  int*   tidx    = (int*)alloc((size_t)B * M);
  // refine-stage f16 buffers overlay crh (dead after cn): need 3*B*M*H = 50.3M
  // halfs out of B*K*H = 67.1M halfs available.
  halft* csubh = crh;                          // B*M*H
  halft* kp2h  = crh + (size_t)B * M * H;      // B*M*H
  halft* vp2h  = crh + (size_t)2 * B * M * H;  // B*M*H

  const float* Wq = ca_in_w;                 const float* bq = ca_in_b;
  const float* Wk = ca_in_w + (size_t)H * H; const float* bk = ca_in_b + H;
  const float* Wv = ca_in_w + (size_t)2*H*H; const float* bv = ca_in_b + 2 * H;
  const float* rWq = rf_in_w;                 const float* rbq = rf_in_b;
  const float* rWk = rf_in_w + (size_t)H * H; const float* rbk = rf_in_b + H;
  const float* rWv = rf_in_w + (size_t)2*H*H; const float* rbv = rf_in_b + 2 * H;

  const float scale = 1.0f / 16.0f;  // 1/sqrt(H)

  // mask = 0
  zero_kernel<<<(B * K + 255) / 256, 256, 0, stream>>>(maskb, B * K);

  // input projections (cr kept in f16 — consumed only as WMMA A operand)
  launch_gemm(query_emb, ipw, (const float*)nullptr, qr, B, D, H, stream);
  launch_gemm(cand_emb,  ipw, (const float*)nullptr, crh, B * K, D, H, stream);

  // cross-attention (K/V projected per batch chunk, stored f16)
  launch_gemm(qr, Wq, bq, qp, B, H, H, stream);
  for (int c = 0; c < B / CH; ++c) {
    const halft* crc = crh + (size_t)c * CH * K * H;
    launch_gemm(crc, Wk, bk, kpch, CH * K, H, H, stream);
    launch_gemm(crc, Wv, bv, vpch, CH * K, H, H, stream);
    attn_kernel<<<CH / 8, 256, 0, stream>>>(qp, kpch, vpch, ctx, c * CH, K, scale);
  }
  launch_gemm(ctx, ca_out_w, ca_out_b, attn_out, B, H, H, stream);
  resln_kernel<<<B / 8, 256, 0, stream>>>(attn_out, qr, ln1_g, ln1_b, qa);

  // h = l2n(qa @ query_proj^T) ; cn = l2n(cr @ cand_proj^T)  (cn stays f32)
  launch_gemm(qa, qpw, (const float*)nullptr, hbuf, B, H, H, stream);
  l2n_kernel<<<B / 8, 256, 0, stream>>>(hbuf);
  launch_gemm(crh, cpw, (const float*)nullptr, cn, B * K, H, H, stream);
  l2n_kernel<<<(B * K) / 8, 256, 0, stream>>>(cn);
  // crh is dead from here on; csubh/kp2h/vp2h alias it.

  for (int s = 0; s < S; ++s) {
    cheap_topk_kernel<<<B / 8, 256, 0, stream>>>(hbuf, step_emb, cn, maskb,
                                                 hs, tidx, csubh, s);
    launch_gemm(hs, rWq, rbq, qp2, B, H, H, stream);
    launch_gemm(csubh, rWk, rbk, kp2h, B * M, H, H, stream);
    launch_gemm(csubh, rWv, rbv, vp2h, B * M, H, H, stream);
    attn_kernel<<<B / 8, 256, 0, stream>>>(qp2, kp2h, vp2h, ctx2, 0, M, scale);
    launch_gemm(ctx2, rf_out_w, rf_out_b, aout2, B, H, H, stream);
    resln_kernel<<<B / 8, 256, 0, stream>>>(aout2, hs, rf_ln_g, rf_ln_b, rtmp);
    l2n_kernel<<<B / 8, 256, 0, stream>>>(rtmp);  // rq
    score_kernel<<<B / 8, 256, 0, stream>>>(rtmp, csubh, tidx, maskb, labels,
                                            cn, logits, preds, chosen, s);
    launch_gemm(chosen, gru_w_ih, gru_b_ih, gi, B, H, 3 * H, stream);
    launch_gemm(hbuf,   gru_w_hh, gru_b_hh, gh, B, H, 3 * H, stream);
    gru_kernel<<<B / 8, 256, 0, stream>>>(gi, gh, hbuf);
  }

  rowloss_kernel<<<(B * S) / 8, 256, 0, stream>>>(logits, labels, rowloss);
  finloss_kernel<<<1, 256, 0, stream>>>(rowloss, lossp);
}
